// MoEForMultiModel_4389456577068
// MI455X (gfx1250) — compile-verified, matrix operand order checked by reading, not run on hardware
//
#include <hip/hip_runtime.h>
#include <hip/hip_bf16.h>

// ---------------------------------------------------------------------------
// MI455X (gfx1250): bf16 WMMA GEMMs + flash attention, async global->LDS DMA.
// Dims from reference: B=4096, DIN=1536, H=1024, NH=8 (hd=128), E=8.
// ---------------------------------------------------------------------------

#define BB   4096
#define DIN  1536
#define HH   1024
#define NHEAD 8
#define HDIM 128
#define NEXP 8

typedef __bf16 v16bf  __attribute__((ext_vector_type(16)));
typedef float  v8f    __attribute__((ext_vector_type(8)));
typedef __bf16 bf16x4 __attribute__((ext_vector_type(4)));
typedef __bf16 bf16x8 __attribute__((ext_vector_type(8)));
typedef int    v4i    __attribute__((ext_vector_type(4)));

// CDNA5 async global->LDS copy path (ASYNCcnt-tracked), guarded so the file
// compiles on toolchains without the builtin (and in the host pass).
// Probe-discovered signature: (v4i* global_src, v4i* lds_dst, imm offset, imm cpol)
#if defined(__AMDGCN__) && __has_builtin(__builtin_amdgcn_global_load_async_to_lds_b128)
#define HAVE_ASYNC_LDS 1
#else
#define HAVE_ASYNC_LDS 0
#endif

__device__ __forceinline__ void wait_async0() {
#if HAVE_ASYNC_LDS
#if __has_builtin(__builtin_amdgcn_s_wait_asynccnt)
  __builtin_amdgcn_s_wait_asynccnt(0);
#else
  asm volatile("s_wait_asynccnt 0x0" ::: "memory");
#endif
#endif
}

#if HAVE_ASYNC_LDS
#define ASYNC_B128(gptr, lptr)                                                 \
  __builtin_amdgcn_global_load_async_to_lds_b128((v4i*)(gptr), (v4i*)(lptr), 0, 0)
#endif

// ---------------------------------------------------------------------------
// Fragment assembly from a row-major LDS tile, matching the CDNA5 ISA layout
// for 16-bit A (16x32) fragments:
//   lane L: m = L&15, half = L>>4
//   element e (0..15): K = (e>>3)*16 + half*8 + (e&7)
// Elements 0..7 and 8..15 are each K-contiguous -> two 16-byte LDS loads.
// Same mapping serves the B (32x16) fragment with n = L&15 when the K-major
// operand is stored as [N][K] row-major (weights / K-tile / V^T).
// ---------------------------------------------------------------------------
__device__ __forceinline__ v16bf make_frag(const __bf16* tile, int ld) {
  int lane = threadIdx.x & 31;
  int m    = lane & 15;
  int half = lane >> 4;
  const bf16x8 lo = *(const bf16x8*)(tile + (long)m * ld + half * 8);
  const bf16x8 hi = *(const bf16x8*)(tile + (long)m * ld + 16 + half * 8);
  v16bf f;
#pragma unroll
  for (int i = 0; i < 8; ++i) { f[i] = lo[i]; f[i + 8] = hi[i]; }
  return f;
}

__device__ __forceinline__ float gelu_exact(float x) {
  return 0.5f * x * (1.0f + erff(x * 0.70710678118654752440f));
}

// ---------------------------------------------------------------------------
// fp32 -> bf16 conversion
// ---------------------------------------------------------------------------
__global__ void cvt_f32_bf16(const float* __restrict__ in, __bf16* __restrict__ out, long n) {
  long i = (long)blockIdx.x * blockDim.x + threadIdx.x;
  long stride = (long)gridDim.x * blockDim.x;
  for (; i < n; i += stride) out[i] = (__bf16)in[i];
}

// ---------------------------------------------------------------------------
// Generic batched GEMM:  C[z] = act( A[z] @ W[z]^T + bias[z] )
//   A: [M,K] bf16 row-major, W: [N,K] bf16 row-major, bias: [N] f32
// Block = 128 threads (4 waves), tile 64x64, K-step 32, double-buffered LDS.
// Wave grid 2x2; each wave owns a 32x32 sub-tile (2x2 WMMA accumulators).
// Tile k+1 is DMA'd (async-to-LDS) while tile k runs on the matrix pipe.
// ---------------------------------------------------------------------------
#define TM 64
#define TN 64
#define TK 32

__global__ void __launch_bounds__(128)
gemm_bf16_wmma(const __bf16* __restrict__ A, const __bf16* __restrict__ W,
               const float* __restrict__ bias,
               float* __restrict__ Cf, __bf16* __restrict__ Cb,
               int M, int N, int K,
               long bsA, long bsW, long bsBias, long bsC, int act) {
  __shared__ __align__(16) __bf16 As[2][TM][TK];
  __shared__ __align__(16) __bf16 Ws[2][TN][TK];

  const int z = blockIdx.z;
  A    += (long)z * bsA;
  W    += (long)z * bsW;
  bias += (long)z * bsBias;
  const long cOff = (long)z * bsC;

  const int bm  = blockIdx.y * TM;
  const int bn  = blockIdx.x * TN;
  const int tid = threadIdx.x;
  const int w    = tid >> 5;
  const int lane = tid & 31;
  const int wm = (w >> 1) * 32;
  const int wn = (w & 1) * 32;

  // b128 staging: 64x32 tile = 256 16-byte chunks, 2 per thread.
  const int cr0 = tid >> 2,         cc0 = (tid & 3) * 8;
  const int cr1 = (tid + 128) >> 2, cc1 = ((tid + 128) & 3) * 8;

  v8f acc[2][2] = {};

#if HAVE_ASYNC_LDS
  auto issue_stage = [&](int k0, int b) {
    ASYNC_B128(&A[(long)(bm + cr0) * K + k0 + cc0], &As[b][cr0][cc0]);
    ASYNC_B128(&A[(long)(bm + cr1) * K + k0 + cc1], &As[b][cr1][cc1]);
    ASYNC_B128(&W[(long)(bn + cr0) * K + k0 + cc0], &Ws[b][cr0][cc0]);
    ASYNC_B128(&W[(long)(bn + cr1) * K + k0 + cc1], &Ws[b][cr1][cc1]);
  };
#else
  bf16x8 ra0, ra1, rw0, rw1;
  auto load_regs = [&](int k0) {
    ra0 = *(const bf16x8*)&A[(long)(bm + cr0) * K + k0 + cc0];
    ra1 = *(const bf16x8*)&A[(long)(bm + cr1) * K + k0 + cc1];
    rw0 = *(const bf16x8*)&W[(long)(bn + cr0) * K + k0 + cc0];
    rw1 = *(const bf16x8*)&W[(long)(bn + cr1) * K + k0 + cc1];
  };
  auto store_regs = [&](int b) {
    *(bf16x8*)&As[b][cr0][cc0] = ra0;
    *(bf16x8*)&As[b][cr1][cc1] = ra1;
    *(bf16x8*)&Ws[b][cr0][cc0] = rw0;
    *(bf16x8*)&Ws[b][cr1][cc1] = rw1;
  };
#endif

  // ---- prologue: stage tile 0 into buffer 0 ----
#if HAVE_ASYNC_LDS
  issue_stage(0, 0);
  wait_async0();
#else
  load_regs(0);
  store_regs(0);
#endif
  __syncthreads();

  int cur = 0;
  for (int k0 = 0; k0 < K; k0 += TK) {
    const int nxt = cur ^ 1;
    const bool more = (k0 + TK) < K;
#if HAVE_ASYNC_LDS
    if (more) issue_stage(k0 + TK, nxt);     // DMA overlaps WMMA below
#else
    if (more) load_regs(k0 + TK);            // loads in flight during WMMA
#endif
    if (k0 + 2 * TK < K) {                   // cache hint for tile after next
      __builtin_prefetch(&A[(long)(bm + (tid >> 1)) * K + k0 + 2 * TK], 0, 1);
      __builtin_prefetch(&W[(long)(bn + (tid >> 1)) * K + k0 + 2 * TK], 0, 1);
    }

    v16bf af[2], wf[2];
#pragma unroll
    for (int t = 0; t < 2; ++t) af[t] = make_frag(&As[cur][wm + t * 16][0], TK);
#pragma unroll
    for (int t = 0; t < 2; ++t) wf[t] = make_frag(&Ws[cur][wn + t * 16][0], TK);
#pragma unroll
    for (int mi = 0; mi < 2; ++mi)
#pragma unroll
      for (int ni = 0; ni < 2; ++ni)
        acc[mi][ni] = __builtin_amdgcn_wmma_f32_16x16x32_bf16(
            false, af[mi], false, wf[ni], (short)0, acc[mi][ni], false, false);

#if HAVE_ASYNC_LDS
    if (more) wait_async0();
#else
    if (more) store_regs(nxt);
#endif
    __syncthreads();
    cur = nxt;
  }

  // Epilogue. C/D layout: lane L -> n = L&15; VGPR r -> m = r + 8*(L>>4).
  const int n_   = lane & 15;
  const int half = lane >> 4;
#pragma unroll
  for (int mi = 0; mi < 2; ++mi)
#pragma unroll
    for (int ni = 0; ni < 2; ++ni)
#pragma unroll
      for (int r = 0; r < 8; ++r) {
        int row = bm + wm + mi * 16 + r + 8 * half;
        int col = bn + wn + ni * 16 + n_;
        float v = acc[mi][ni][r] + bias[col];
        if (act) v = gelu_exact(v);
        long idx = cOff + (long)row * N + col;
        if (Cf) Cf[idx] = v;
        if (Cb) Cb[idx] = (__bf16)v;
      }
}

// ---------------------------------------------------------------------------
// Flash attention over the batch axis (seq len = B = 4096, 8 heads, hd=128).
// qkv: [B, 3*H] bf16 (q | k | v; head slice h*128..h*128+127 in each third)
// out: [B, H]   bf16
// Grid: (B/64, NH). Block: 256 threads = 8 waves (4x2 wave grid).
//   S tile 64x64: wave (wm,wn) computes rows wm*16..+16, cols wn*32..+32.
//   O tile 64x128: wave computes rows wm*16..+16, cols wn*64..+64.
// K is DMA'd async to LDS; V global loads are issued before the K wait so
// their latency hides under the S WMMAs + softmax.
// ---------------------------------------------------------------------------
struct __align__(16) FlashSmem {
  __bf16 Qs[64][128];                               // 16 KB, persistent
  union { __bf16 Ks[64][128]; __bf16 Vt[128][64]; };// 16 KB, phase-exclusive
  float  Ss[64][64];                                // 16 KB raw scores
  __bf16 Ps[64][64];                                //  8 KB probabilities
  float  mrow[64];
  float  lrow[64];
  float  arow[64];
};

__global__ void __launch_bounds__(256)
flash_attn_bf16(const __bf16* __restrict__ qkv, __bf16* __restrict__ out, int Bn) {
  __shared__ FlashSmem sm;
  const int i0  = blockIdx.x * 64;
  const int hh  = blockIdx.y;
  const int tid = threadIdx.x;
  const int w    = tid >> 5;
  const int lane = tid & 31;
  const int wm = w >> 1;   // 0..3
  const int wn = w & 1;    // 0..1
  const long rowStride = 3 * HH;
  const int  hOff = hh * HDIM;
  const float scale = 0.08838834764831845f;  // 1/sqrt(128)

  // b128 staging of a 64x128 tile: 1024 chunks, 4 per thread.
  // chunk c: row = c>>4, col = (c&15)*8
#if HAVE_ASYNC_LDS
#pragma unroll
  for (int i = 0; i < 4; ++i) {
    int c = tid + i * 256, r = c >> 4, col = (c & 15) * 8;
    ASYNC_B128(&qkv[(long)(i0 + r) * rowStride + hOff + col], &sm.Qs[r][col]);
  }
#else
  {
    bf16x8 q[4];
#pragma unroll
    for (int i = 0; i < 4; ++i) {
      int c = tid + i * 256, r = c >> 4, col = (c & 15) * 8;
      q[i] = *(const bf16x8*)&qkv[(long)(i0 + r) * rowStride + hOff + col];
    }
#pragma unroll
    for (int i = 0; i < 4; ++i) {
      int c = tid + i * 256, r = c >> 4, col = (c & 15) * 8;
      *(bf16x8*)&sm.Qs[r][col] = q[i];
    }
  }
#endif
  if (tid < 64) { sm.mrow[tid] = -1e30f; sm.lrow[tid] = 0.0f; }

  v8f accO[4] = {};

  for (int j0 = 0; j0 < Bn; j0 += 64) {
    // ---- stage K tile (64x128) ----
#if HAVE_ASYNC_LDS
#pragma unroll
    for (int i = 0; i < 4; ++i) {
      int c = tid + i * 256, r = c >> 4, col = (c & 15) * 8;
      ASYNC_B128(&qkv[(long)(j0 + r) * rowStride + HH + hOff + col], &sm.Ks[r][col]);
    }
#else
    bf16x8 kr[4];
#pragma unroll
    for (int i = 0; i < 4; ++i) {
      int c = tid + i * 256, r = c >> 4, col = (c & 15) * 8;
      kr[i] = *(const bf16x8*)&qkv[(long)(j0 + r) * rowStride + HH + hOff + col];
    }
#endif
    // ---- issue V loads into registers now; latency hides under S/softmax ----
    bf16x8 vr[4];
#pragma unroll
    for (int i = 0; i < 4; ++i) {
      int c = tid + i * 256, r = c >> 4, col = (c & 15) * 8;
      vr[i] = *(const bf16x8*)&qkv[(long)(j0 + r) * rowStride + 2 * HH + hOff + col];
    }
#if HAVE_ASYNC_LDS
    wait_async0();
#else
#pragma unroll
    for (int i = 0; i < 4; ++i) {
      int c = tid + i * 256, r = c >> 4, col = (c & 15) * 8;
      *(bf16x8*)&sm.Ks[r][col] = kr[i];
    }
#endif
    __syncthreads();

    // ---- S = Q @ K^T (contraction over d = 128, 4 WMMA k-steps) ----
    v8f accS[2] = {};
#pragma unroll
    for (int kk = 0; kk < HDIM; kk += 32) {
      v16bf aq = make_frag(&sm.Qs[wm * 16][kk], HDIM);
#pragma unroll
      for (int nt = 0; nt < 2; ++nt) {
        v16bf bk = make_frag(&sm.Ks[wn * 32 + nt * 16][kk], HDIM);
        accS[nt] = __builtin_amdgcn_wmma_f32_16x16x32_bf16(
            false, aq, false, bk, (short)0, accS[nt], false, false);
      }
    }
    {
      const int n_ = lane & 15, half = lane >> 4;
#pragma unroll
      for (int nt = 0; nt < 2; ++nt)
#pragma unroll
        for (int r = 0; r < 8; ++r)
          sm.Ss[wm * 16 + r + 8 * half][wn * 32 + nt * 16 + n_] = accS[nt][r] * scale;
    }
    __syncthreads();

    // ---- online softmax: one thread per row ----
    if (tid < 64) {
      float mOld = sm.mrow[tid];
      float mNew = mOld;
#pragma unroll 8
      for (int j = 0; j < 64; ++j) mNew = fmaxf(mNew, sm.Ss[tid][j]);
      float alpha = __expf(mOld - mNew);
      float lsum = 0.0f;
      for (int j = 0; j < 64; ++j) {
        float p = __expf(sm.Ss[tid][j] - mNew);
        sm.Ps[tid][j] = (__bf16)p;
        lsum += p;
      }
      sm.mrow[tid] = mNew;
      sm.lrow[tid] = sm.lrow[tid] * alpha + lsum;
      sm.arow[tid] = alpha;
    }
    __syncthreads();

    // ---- rescale O accumulators, scatter V^T into (dead) K region ----
    {
      const int half = lane >> 4;
#pragma unroll
      for (int nt = 0; nt < 4; ++nt)
#pragma unroll
        for (int r = 0; r < 8; ++r)
          accO[nt][r] *= sm.arow[wm * 16 + r + 8 * half];
    }
#pragma unroll
    for (int i = 0; i < 4; ++i) {
      int c = tid + i * 256, r = c >> 4, col = (c & 15) * 8;
#pragma unroll
      for (int p = 0; p < 8; ++p) sm.Vt[col + p][r] = vr[i][p];
    }
    __syncthreads();

    // ---- O += P @ V (contraction over j = 64, 2 WMMA k-steps) ----
#pragma unroll
    for (int kk = 0; kk < 64; kk += 32) {
      v16bf ap = make_frag(&sm.Ps[wm * 16][kk], 64);
#pragma unroll
      for (int nt = 0; nt < 4; ++nt) {
        v16bf bv = make_frag(&sm.Vt[wn * 64 + nt * 16][kk], 64);
        accO[nt] = __builtin_amdgcn_wmma_f32_16x16x32_bf16(
            false, ap, false, bv, (short)0, accO[nt], false, false);
      }
    }
    __syncthreads();
  }

  // ---- normalize and write O ----
  {
    const int n_ = lane & 15, half = lane >> 4;
#pragma unroll
    for (int nt = 0; nt < 4; ++nt)
#pragma unroll
      for (int r = 0; r < 8; ++r) {
        int row = wm * 16 + r + 8 * half;
        int col = wn * 64 + nt * 16 + n_;
        float v = accO[nt][r] / sm.lrow[row];
        out[(long)(i0 + row) * HH + hOff + col] = (__bf16)v;
      }
  }
}

// ---------------------------------------------------------------------------
// LayerNorm over H=1024; one block per row; writes f32 (for gate) + bf16.
// ---------------------------------------------------------------------------
__global__ void __launch_bounds__(256)
layernorm_k(const float* __restrict__ in, const float* __restrict__ g,
            const float* __restrict__ b, float* __restrict__ outF,
            __bf16* __restrict__ outB, int Hn) {
  __shared__ float red[256];
  const int row = blockIdx.x;
  const float* x = in + (long)row * Hn;
  float s = 0.0f;
  for (int i = threadIdx.x; i < Hn; i += 256) s += x[i];
  red[threadIdx.x] = s; __syncthreads();
  for (int st = 128; st > 0; st >>= 1) {
    if (threadIdx.x < st) red[threadIdx.x] += red[threadIdx.x + st];
    __syncthreads();
  }
  float mu = red[0] / Hn;
  __syncthreads();
  float vs = 0.0f;
  for (int i = threadIdx.x; i < Hn; i += 256) { float d = x[i] - mu; vs += d * d; }
  red[threadIdx.x] = vs; __syncthreads();
  for (int st = 128; st > 0; st >>= 1) {
    if (threadIdx.x < st) red[threadIdx.x] += red[threadIdx.x + st];
    __syncthreads();
  }
  float rstd = rsqrtf(red[0] / Hn + 1e-5f);
  for (int i = threadIdx.x; i < Hn; i += 256) {
    float v = (x[i] - mu) * rstd * g[i] + b[i];
    outF[(long)row * Hn + i] = v;
    outB[(long)row * Hn + i] = (__bf16)v;
  }
}

// ---------------------------------------------------------------------------
// Gate: softmax(h @ gate_W^T + gate_b) -> top-2, renormalized.
// ---------------------------------------------------------------------------
__global__ void __launch_bounds__(256)
gate_top2_k(const float* __restrict__ h, const float* __restrict__ gw,
            const float* __restrict__ gb, float* __restrict__ topv,
            int* __restrict__ topi, int Hn) {
  __shared__ float red[256];
  __shared__ float logits[NEXP];
  const int row = blockIdx.x;
  const float* x = h + (long)row * Hn;
  for (int e = 0; e < NEXP; ++e) {
    float s = 0.0f;
    for (int i = threadIdx.x; i < Hn; i += 256) s += x[i] * gw[(long)e * Hn + i];
    red[threadIdx.x] = s; __syncthreads();
    for (int st = 128; st > 0; st >>= 1) {
      if (threadIdx.x < st) red[threadIdx.x] += red[threadIdx.x + st];
      __syncthreads();
    }
    if (threadIdx.x == 0) logits[e] = red[0] + gb[e];
    __syncthreads();
  }
  if (threadIdx.x == 0) {
    float mx = logits[0];
    for (int e = 1; e < NEXP; ++e) mx = fmaxf(mx, logits[e]);
    float p[NEXP], den = 0.0f;
    for (int e = 0; e < NEXP; ++e) { p[e] = __expf(logits[e] - mx); den += p[e]; }
    int i0 = 0;
    for (int e = 1; e < NEXP; ++e) if (p[e] > p[i0]) i0 = e;
    int i1 = (i0 == 0) ? 1 : 0;
    for (int e = 0; e < NEXP; ++e) if (e != i0 && p[e] > p[i1]) i1 = e;
    float v0 = p[i0] / den, v1 = p[i1] / den, sum = v0 + v1;
    topv[row * 2 + 0] = v0 / sum;
    topv[row * 2 + 1] = v1 / sum;
    topi[row * 2 + 0] = i0;
    topi[row * 2 + 1] = i1;
  }
}

// ---------------------------------------------------------------------------
// Expert layer 5: e5[b][e] = dot(e4[e][b][:128], W5[e]) + b5[e]  (N=1)
// ---------------------------------------------------------------------------
__global__ void expert5_k(const __bf16* __restrict__ e4, const float* __restrict__ W5,
                          const float* __restrict__ b5, float* __restrict__ e5, int Bn) {
  int idx = blockIdx.x * blockDim.x + threadIdx.x;
  if (idx >= Bn * NEXP) return;
  int b = idx >> 3, e = idx & 7;
  const __bf16* a = e4 + ((long)e * Bn + b) * HDIM;
  const float* wv = W5 + e * HDIM;
  float s = 0.0f;
#pragma unroll 16
  for (int r = 0; r < HDIM; ++r) s += (float)a[r] * wv[r];
  e5[(long)b * NEXP + e] = s + b5[e];
}

// ---------------------------------------------------------------------------
// Final: out[b] = sigmoid( sum_k topv * e5[topi] )
// ---------------------------------------------------------------------------
__global__ void final_k(const float* __restrict__ e5, const float* __restrict__ topv,
                        const int* __restrict__ topi, float* __restrict__ out, int Bn) {
  int b = blockIdx.x * blockDim.x + threadIdx.x;
  if (b >= Bn) return;
  float s = topv[b * 2 + 0] * e5[(long)b * NEXP + topi[b * 2 + 0]] +
            topv[b * 2 + 1] * e5[(long)b * NEXP + topi[b * 2 + 1]];
  out[b] = 1.0f / (1.0f + __expf(-s));
}

// ---------------------------------------------------------------------------
// Host orchestration
// ---------------------------------------------------------------------------
static inline dim3 gemm_grid(int M, int N, int batch) {
  return dim3((unsigned)(N / TN), (unsigned)(M / TM), (unsigned)batch);
}

extern "C" void kernel_launch(void* const* d_in, const int* in_sizes, int n_in,
                              void* d_out, int out_size, void* d_ws, size_t ws_size,
                              hipStream_t stream) {
  (void)in_sizes; (void)n_in; (void)out_size; (void)ws_size;

  const float* x        = (const float*)d_in[0];
  const float* proj_W   = (const float*)d_in[1];
  const float* proj_b   = (const float*)d_in[2];
  const float* inW      = (const float*)d_in[3];
  const float* inB      = (const float*)d_in[4];
  const float* outW     = (const float*)d_in[5];
  const float* outBias  = (const float*)d_in[6];
  const float* ln_g     = (const float*)d_in[7];
  const float* ln_b     = (const float*)d_in[8];
  const float* gate_W   = (const float*)d_in[9];
  const float* gate_b   = (const float*)d_in[10];
  const float* W1 = (const float*)d_in[11]; const float* b1 = (const float*)d_in[12];
  const float* W2 = (const float*)d_in[13]; const float* b2 = (const float*)d_in[14];
  const float* W3 = (const float*)d_in[15]; const float* b3 = (const float*)d_in[16];
  const float* W4 = (const float*)d_in[17]; const float* b4 = (const float*)d_in[18];
  const float* W5 = (const float*)d_in[19]; const float* b5 = (const float*)d_in[20];
  float* out = (float*)d_out;

  // ---- workspace partition ----
  char* base = (char*)d_ws;
  size_t off = 0;
  auto take = [&](size_t bytes) -> char* {
    char* p = base + off;
    off = (off + bytes + 255) & ~(size_t)255;
    return p;
  };
  __bf16* xb    = (__bf16*)take((size_t)BB * DIN * 2);
  __bf16* pWb   = (__bf16*)take((size_t)HH * DIN * 2);
  __bf16* ipWb  = (__bf16*)take((size_t)3 * HH * HH * 2);
  __bf16* opWb  = (__bf16*)take((size_t)HH * HH * 2);
  __bf16* W1b   = (__bf16*)take((size_t)NEXP * 1024 * HH * 2);
  __bf16* W2b   = (__bf16*)take((size_t)NEXP * 512 * 1024 * 2);
  __bf16* W3b   = (__bf16*)take((size_t)NEXP * 256 * 512 * 2);
  __bf16* W4b   = (__bf16*)take((size_t)NEXP * 128 * 256 * 2);
  __bf16* projb = (__bf16*)take((size_t)BB * HH * 2);
  __bf16* qkvb  = (__bf16*)take((size_t)BB * 3 * HH * 2);
  __bf16* aob   = (__bf16*)take((size_t)BB * HH * 2);
  float*  aof   = (float*) take((size_t)BB * HH * 4);
  float*  hf    = (float*) take((size_t)BB * HH * 4);
  __bf16* hb    = (__bf16*)take((size_t)BB * HH * 2);
  __bf16* e1b   = (__bf16*)take((size_t)NEXP * BB * 1024 * 2);
  __bf16* e2b   = (__bf16*)take((size_t)NEXP * BB * 512 * 2);
  __bf16* e3b   = (__bf16*)take((size_t)NEXP * BB * 256 * 2);
  __bf16* e4b   = (__bf16*)take((size_t)NEXP * BB * 128 * 2);
  float*  e5f   = (float*) take((size_t)BB * NEXP * 4);
  float*  topv  = (float*) take((size_t)BB * 2 * 4);
  int*    topi  = (int*)   take((size_t)BB * 2 * 4);

  auto cvt = [&](const float* src, __bf16* dst, long n) {
    long blocks = (n + 255) / 256;
    if (blocks > 4096) blocks = 4096;
    cvt_f32_bf16<<<(unsigned)blocks, 256, 0, stream>>>(src, dst, n);
  };

  // ---- fp32 -> bf16 casts (x + weights) ----
  cvt(x,      xb,   (long)BB * DIN);
  cvt(proj_W, pWb,  (long)HH * DIN);
  cvt(inW,    ipWb, (long)3 * HH * HH);
  cvt(outW,   opWb, (long)HH * HH);
  cvt(W1,     W1b,  (long)NEXP * 1024 * HH);
  cvt(W2,     W2b,  (long)NEXP * 512 * 1024);
  cvt(W3,     W3b,  (long)NEXP * 256 * 512);
  cvt(W4,     W4b,  (long)NEXP * 128 * 256);

  // ---- proj = x @ proj_W^T + proj_b  -> bf16 ----
  gemm_bf16_wmma<<<gemm_grid(BB, HH, 1), 128, 0, stream>>>(
      xb, pWb, proj_b, nullptr, projb, BB, HH, DIN, 0, 0, 0, 0, 0);

  // ---- qkv = proj @ in_proj_W^T + in_proj_b -> bf16 ----
  gemm_bf16_wmma<<<gemm_grid(BB, 3 * HH, 1), 128, 0, stream>>>(
      projb, ipWb, inB, nullptr, qkvb, BB, 3 * HH, HH, 0, 0, 0, 0, 0);

  // ---- flash attention over the batch axis ----
  flash_attn_bf16<<<dim3(BB / 64, NHEAD), 256, 0, stream>>>(qkvb, aob, BB);

  // ---- out_proj -> f32 (feeds LayerNorm) ----
  gemm_bf16_wmma<<<gemm_grid(BB, HH, 1), 128, 0, stream>>>(
      aob, opWb, outBias, aof, nullptr, BB, HH, HH, 0, 0, 0, 0, 0);

  // ---- LayerNorm ----
  layernorm_k<<<BB, 256, 0, stream>>>(aof, ln_g, ln_b, hf, hb, HH);

  // ---- gate + top-2 ----
  gate_top2_k<<<BB, 256, 0, stream>>>(hf, gate_W, gate_b, topv, topi, HH);

  // ---- experts, batched over E via gridDim.z ----
  gemm_bf16_wmma<<<gemm_grid(BB, 1024, NEXP), 128, 0, stream>>>(
      hb, W1b, b1, nullptr, e1b, BB, 1024, HH,
      0, (long)1024 * HH, 1024, (long)BB * 1024, 1);
  gemm_bf16_wmma<<<gemm_grid(BB, 512, NEXP), 128, 0, stream>>>(
      e1b, W2b, b2, nullptr, e2b, BB, 512, 1024,
      (long)BB * 1024, (long)512 * 1024, 512, (long)BB * 512, 1);
  gemm_bf16_wmma<<<gemm_grid(BB, 256, NEXP), 128, 0, stream>>>(
      e2b, W3b, b3, nullptr, e3b, BB, 256, 512,
      (long)BB * 512, (long)256 * 512, 256, (long)BB * 256, 1);
  gemm_bf16_wmma<<<gemm_grid(BB, 128, NEXP), 128, 0, stream>>>(
      e3b, W4b, b4, nullptr, e4b, BB, 128, 256,
      (long)BB * 256, (long)128 * 256, 128, (long)BB * 128, 1);

  // ---- expert head (N=1) + gather + sigmoid ----
  expert5_k<<<(BB * NEXP + 255) / 256, 256, 0, stream>>>(e4b, W5, b5, e5f, BB);
  final_k<<<(BB + 255) / 256, 256, 0, stream>>>(e5f, topv, topi, out, BB);
}
